// JODIO_51393578663967
// MI455X (gfx1250) — matrix operand order repaired and checked by reading, not compiled, not run on hardware
//
#include <hip/hip_runtime.h>
#include <stdint.h>

// ---------------------------------------------------------------------------
// Residual Vector Quantizer for MI455X (gfx1250), wave32 + WMMA bf16 path.
// v2: workgroup-cooperative B-tile staging through LDS using gfx1250 async
//     global->LDS copies (ASYNCcnt) with double buffering. Cuts L2 codebook
//     traffic 8x (each tile loaded once per WG instead of once per wave).
//
// B=8, L=8192, D=128, K=1024, Q=8.
// scores = |c|^2 - 2 r.c  -> argmin over K, gather, residual update, x8 levels
// ---------------------------------------------------------------------------

typedef __attribute__((ext_vector_type(16))) __bf16 v16bf;
typedef __attribute__((ext_vector_type(8)))  float  v8f;
typedef __attribute__((ext_vector_type(4)))  float  v4f;

#define QQ    8
#define KK    1024
#define DDIM  128
#define NROWS 65536            // B*L
#define WAVES_PER_WG 8
#define TILE_BYTES   4096      // 16 codes * 128 * 2B (one bf16 part)

union Vb {
    v16bf v;
    uint4 q[2];
    unsigned short s[16];
};

__device__ __forceinline__ unsigned short f2bf(float f) {
    union { float f; unsigned u; } c; c.f = f;
    unsigned u = c.u;
    unsigned r = u + 0x7FFFu + ((u >> 16) & 1u);   // RNE truncate to bf16
    return (unsigned short)(r >> 16);
}
__device__ __forceinline__ float bf2f(unsigned short h) {
    union { unsigned u; float f; } c; c.u = ((unsigned)h) << 16;
    return c.f;
}

// gfx1250 async global->LDS copy, 16 bytes per lane. Tracked by ASYNCcnt.
__device__ __forceinline__ void async_copy_b128(uint32_t lds_off, uint64_t gaddr) {
    asm volatile("global_load_async_to_lds_b128 %0, %1, off"
                 :: "v"(lds_off), "v"(gaddr) : "memory");
}
__device__ __forceinline__ void wait_async0() {
#if __has_builtin(__builtin_amdgcn_s_wait_asynccnt)
    __builtin_amdgcn_s_wait_asynccnt(0);
#else
    asm volatile("s_wait_asynccnt 0x0" ::: "memory");
#endif
}

// --------------------------- prep kernels ----------------------------------

// Split f32 codebooks into bf16 hi/lo (elementwise).
__global__ void rvq_split_cb(const float* __restrict__ cb,
                             unsigned short* __restrict__ hi,
                             unsigned short* __restrict__ lo, int n) {
    int i = blockIdx.x * blockDim.x + threadIdx.x;
    if (i < n) {
        float v = cb[i];
        unsigned short h = f2bf(v);
        hi[i] = h;
        lo[i] = f2bf(v - bf2f(h));
    }
}

// Per-code squared norm |c|^2 (one thread per code; 8192 codes total).
__global__ void rvq_c2(const float* __restrict__ cb, float* __restrict__ c2) {
    int code = blockIdx.x * blockDim.x + threadIdx.x;
    if (code < QQ * KK) {
        const float* p = cb + (size_t)code * DDIM;
        float s = 0.f;
        #pragma unroll 16
        for (int d = 0; d < DDIM; ++d) s += p[d] * p[d];
        c2[code] = s;
    }
}

// ----------------------------- main kernel ---------------------------------

__global__ __launch_bounds__(32 * WAVES_PER_WG)
void rvq_main(const float* __restrict__ x,
              const float* __restrict__ cb,            // f32 [Q,K,D]
              const unsigned short* __restrict__ cb_hi,// bf16 [Q,K,D]
              const unsigned short* __restrict__ cb_lo,// bf16 [Q,K,D]
              const float* __restrict__ c2t,           // [Q,K]
              float* __restrict__ out_idx,             // [Q, B*L] (indices as float)
              float* __restrict__ out_q)               // [B*L, D]
{
    // Double-buffered B tile stage: [buf][hi/lo][16 codes * 128 elems]
    __shared__ __align__(16) unsigned short lds_b[2][2][16 * DDIM];

    const int tid  = threadIdx.x;
    const int lane = tid & 31;
    const int wave = tid >> 5;
    const int r0   = (blockIdx.x * WAVES_PER_WG + wave) * 16;  // row-tile base
    const int m    = lane & 15;       // A row / B column owned by this lane
    const int half = lane >> 4;       // which K-subset this lane holds

    const uint32_t lds_base = (uint32_t)(uintptr_t)(&lds_b[0][0][0]);
    const uint32_t my_cp    = (uint32_t)tid * 16;   // 16B chunk per thread

    // Residual in registers, A-operand-friendly layout:
    //   res[c][e]: e in 0..7  -> k = c*32 + half*8 + e
    //              e in 8..15 -> k = c*32 + 16 + half*8 + (e-8)
    float res[4][16];
    const float* xrow = x + (size_t)(r0 + m) * DDIM;
    #pragma unroll
    for (int c = 0; c < 4; ++c) {
        const int k0 = c * 32 + half * 8;
        v4f a0 = *(const v4f*)(xrow + k0);
        v4f a1 = *(const v4f*)(xrow + k0 + 4);
        v4f a2 = *(const v4f*)(xrow + k0 + 16);
        v4f a3 = *(const v4f*)(xrow + k0 + 20);
        #pragma unroll
        for (int i = 0; i < 4; ++i) {
            res[c][i]      = a0[i];
            res[c][4 + i]  = a1[i];
            res[c][8 + i]  = a2[i];
            res[c][12 + i] = a3[i];
        }
    }

    for (int q = 0; q < QQ; ++q) {
        const unsigned short* cbh = cb_hi + (size_t)q * KK * DDIM;
        const unsigned short* cbl = cb_lo + (size_t)q * KK * DDIM;
        const float*          c2q = c2t + q * KK;
        const uint64_t gh = (uint64_t)(uintptr_t)cbh;
        const uint64_t gl = (uint64_t)(uintptr_t)cbl;

        // ---- stage tile 0 into buffer 0 (async, overlaps the split) ----
        async_copy_b128(lds_base + 0 * 2 * TILE_BYTES + 0          + my_cp,
                        gh + 0 * TILE_BYTES + my_cp);
        async_copy_b128(lds_base + 0 * 2 * TILE_BYTES + TILE_BYTES + my_cp,
                        gl + 0 * TILE_BYTES + my_cp);

        // ---- split residual into bf16 hi/lo WMMA A operands ----
        v16bf a_hi[4], a_lo[4];
        #pragma unroll
        for (int c = 0; c < 4; ++c) {
            Vb th, tl;
            #pragma unroll
            for (int e = 0; e < 16; ++e) {
                float v = res[c][e];
                unsigned short h = f2bf(v);
                th.s[e] = h;
                tl.s[e] = f2bf(v - bf2f(h));
            }
            a_hi[c] = th.v;
            a_lo[c] = tl.v;
        }

        float minv[8];
        int   mini[8];
        #pragma unroll
        for (int r = 0; r < 8; ++r) { minv[r] = 3.4e38f; mini[r] = 0; }

        for (int t = 0; t < 64; ++t) {
            // Tile t is fully in LDS once every wave's own copies completed
            // and all waves hit the barrier.
            wait_async0();
            __syncthreads();

            // Issue next tile into the other buffer (its previous readers
            // finished before the barrier above).
            if (t < 63) {
                const uint32_t nb = (uint32_t)((t + 1) & 1) * (2 * TILE_BYTES);
                const uint64_t go = (uint64_t)(t + 1) * TILE_BYTES;
                async_copy_b128(lds_base + nb + 0          + my_cp, gh + go + my_cp);
                async_copy_b128(lds_base + nb + TILE_BYTES + my_cp, gl + go + my_cp);
            }

            const int buf  = t & 1;
            const int code = t * 16 + m;                 // this lane's B column
            const unsigned short* lh = &lds_b[buf][0][m * DDIM];
            const unsigned short* ll = &lds_b[buf][1][m * DDIM];
            const float c2v = c2q[code];

            v8f acc = {};
            #pragma unroll
            for (int c = 0; c < 4; ++c) {
                // B layout: lane half h holds K = c*32+8h+0..7 and
                // K = c*32+16+8h+0..7 of column `code` (matches A).
                const int kb = c * 32 + half * 8;
                Vb Bh, Bl;
                Bh.q[0] = *(const uint4*)(lh + kb);
                Bh.q[1] = *(const uint4*)(lh + kb + 16);
                Bl.q[0] = *(const uint4*)(ll + kb);
                Bl.q[1] = *(const uint4*)(ll + kb + 16);

                acc = __builtin_amdgcn_wmma_f32_16x16x32_bf16(
                        false, a_hi[c], false, Bh.v, (short)0, acc, false, false);
                acc = __builtin_amdgcn_wmma_f32_16x16x32_bf16(
                        false, a_lo[c], false, Bh.v, (short)0, acc, false, false);
                acc = __builtin_amdgcn_wmma_f32_16x16x32_bf16(
                        false, a_hi[c], false, Bl.v, (short)0, acc, false, false);
            }
            // C layout: this lane holds 8 rows of column `code`.
            #pragma unroll
            for (int r = 0; r < 8; ++r) {
                float s = c2v - 2.0f * acc[r];
                if (s < minv[r]) { minv[r] = s; mini[r] = code; }
            }
        }

        // ---- reduce argmin across the 16 columns in each lane half ----
        #pragma unroll
        for (int r = 0; r < 8; ++r) {
            #pragma unroll
            for (int off = 1; off < 16; off <<= 1) {
                float ov = __shfl_xor(minv[r], off, 32);
                int   oi = __shfl_xor(mini[r], off, 32);
                if (ov < minv[r] || (ov == minv[r] && oi < mini[r])) {
                    minv[r] = ov; mini[r] = oi;
                }
            }
        }
        // lanes 0-15 now hold rows 0..7 in mini[0..7]; lanes 16-31 rows 8..15.
        // Route row (lane&15)'s winner to this lane: select + xor-24 shuffle.
        const int l7 = lane & 7;
        int selLocal = mini[0];
        #pragma unroll
        for (int r = 1; r < 8; ++r) selLocal = (l7 == r) ? mini[r] : selLocal;
        int other = __shfl_xor(selLocal, 24, 32);
        int myidx = ((((lane >> 3) ^ (lane >> 4)) & 1) ? other : selLocal);

        // ---- emit index (one lane per row); EXEC reconverges after ----
        if (half == 0) {
            out_idx[(size_t)q * NROWS + r0 + m] = (float)myidx;
        }

        // ---- gather winning code (f32) and update residual exactly ----
        const float* cr = cb + ((size_t)q * KK + (size_t)myidx) * DDIM;
        #pragma unroll
        for (int c = 0; c < 4; ++c) {
            const int k0 = c * 32 + half * 8;
            v4f q0 = *(const v4f*)(cr + k0);
            v4f q1 = *(const v4f*)(cr + k0 + 4);
            v4f q2 = *(const v4f*)(cr + k0 + 16);
            v4f q3 = *(const v4f*)(cr + k0 + 20);
            #pragma unroll
            for (int i = 0; i < 4; ++i) {
                res[c][i]      -= q0[i];
                res[c][4 + i]  -= q1[i];
                res[c][8 + i]  -= q2[i];
                res[c][12 + i] -= q3[i];
            }
        }

        // All reads of this level's LDS buffers are done before the next
        // level starts overwriting them.
        __syncthreads();
    }

    // quantized = x - final residual (same strided layout)
    float* orow = out_q + (size_t)(r0 + m) * DDIM;
    #pragma unroll
    for (int c = 0; c < 4; ++c) {
        const int k0 = c * 32 + half * 8;
        v4f a0 = *(const v4f*)(xrow + k0);
        v4f a1 = *(const v4f*)(xrow + k0 + 4);
        v4f a2 = *(const v4f*)(xrow + k0 + 16);
        v4f a3 = *(const v4f*)(xrow + k0 + 20);
        v4f o0, o1, o2, o3;
        #pragma unroll
        for (int i = 0; i < 4; ++i) {
            o0[i] = a0[i] - res[c][i];
            o1[i] = a1[i] - res[c][4 + i];
            o2[i] = a2[i] - res[c][8 + i];
            o3[i] = a3[i] - res[c][12 + i];
        }
        *(v4f*)(orow + k0)      = o0;
        *(v4f*)(orow + k0 + 4)  = o1;
        *(v4f*)(orow + k0 + 16) = o2;
        *(v4f*)(orow + k0 + 20) = o3;
    }
}

// ------------------------------ launcher -----------------------------------

extern "C" void kernel_launch(void* const* d_in, const int* in_sizes, int n_in,
                              void* d_out, int out_size, void* d_ws, size_t ws_size,
                              hipStream_t stream) {
    (void)in_sizes; (void)n_in; (void)out_size; (void)ws_size;
    const float* x  = (const float*)d_in[0];   // [B, L, D]
    const float* cb = (const float*)d_in[1];   // [Q, K, D]

    const int QKD = QQ * KK * DDIM;            // 1,048,576 elements

    // workspace: cb_hi (2MB bf16) | cb_lo (2MB bf16) | c2 (32KB f32)
    unsigned short* cb_hi = (unsigned short*)d_ws;
    unsigned short* cb_lo = cb_hi + QKD;
    float*          c2    = (float*)(cb_lo + QKD);

    rvq_split_cb<<<(QKD + 255) / 256, 256, 0, stream>>>(cb, cb_hi, cb_lo, QKD);
    rvq_c2<<<(QQ * KK + 255) / 256, 256, 0, stream>>>(cb, c2);

    float* out_idx = (float*)d_out;                      // [Q, B*L]
    float* out_q   = out_idx + (size_t)QQ * NROWS;       // [B*L, D]

    const int wgs = NROWS / (16 * WAVES_PER_WG);         // 512 workgroups
    rvq_main<<<wgs, 32 * WAVES_PER_WG, 0, stream>>>(
        x, cb, cb_hi, cb_lo, c2, out_idx, out_q);
}
